// EncoderAttDecoder_56642028699772
// MI455X (gfx1250) — matrix operand order repaired and checked by reading, not compiled, not run on hardware
//
#include <hip/hip_runtime.h>
#include <hip/hip_bf16.h>
#include <math.h>

// dims fixed by the reference
#define IN_H    256
#define OUT_H   256
#define DE_H    512
#define TAGS    6
#define MAXLEN  1024
#define START_TAG 4

typedef __attribute__((ext_vector_type(16))) __bf16 v16bf;
typedef __attribute__((ext_vector_type(8)))  __bf16 v8bf;
typedef __attribute__((ext_vector_type(8)))  float  v8f;

__device__ __forceinline__ float sigm(float x) { return 1.0f / (1.0f + expf(-x)); }

// async global->LDS copy of one dword per lane (CDNA5 async-DMA path, ASYNCcnt-tracked)
__device__ __forceinline__ void async_copy_b32(void* lds_dst, const void* gaddr) {
  unsigned lds = (unsigned)(uintptr_t)lds_dst; // low 32 bits of generic shared ptr = LDS offset
  asm volatile("global_load_async_to_lds_b32 %0, %1, off"
               :: "v"(lds), "v"(gaddr) : "memory");
}

// -------------------- small utility kernels --------------------
__global__ void k_f32_to_bf16(const float* __restrict__ s, __bf16* __restrict__ d, int n) {
  int i = blockIdx.x * blockDim.x + threadIdx.x;
  if (i < n) d[i] = (__bf16)s[i];
}

__global__ void k_fill0(float* __restrict__ d, int n) {
  int i = blockIdx.x * blockDim.x + threadIdx.x;
  if (i < n) d[i] = 0.0f;
}

// gather token embeddings -> bf16 activation matrix X [L, IN_H]
__global__ void k_embed(const int* __restrict__ x, const float* __restrict__ emb,
                        __bf16* __restrict__ X, int L) {
  int i = blockIdx.x * blockDim.x + threadIdx.x;
  if (i < L * IN_H) {
    int t = i >> 8;         // / IN_H
    int j = i & (IN_H - 1); // % IN_H
    X[i] = (__bf16)emb[(size_t)x[t] * IN_H + j];
  }
}

// -------------------- WMMA bf16 GEMM: C[M,N] = A[M,K] @ B[N,K]^T + bias --------------------
// One wave computes a 16(M) x 64(N) tile: 4 independent accumulator chains + 4x A reuse.
// Software-pipelined: fragments for k-step k+32 are loaded before the WMMAs of k-step k,
// so WMMAs only take partial loadcnt waits. K is compile-time -> full unroll, copies vanish.
// bias split: col < nsplit -> bias0[col], else bias1[col-nsplit]  (f / b gate stacking)
template <int K>
__global__ __launch_bounds__(32)
void k_wmma_gemm(const __bf16* __restrict__ A, const __bf16* __restrict__ B,
                 const float* __restrict__ bias0, const float* __restrict__ bias1,
                 int nsplit, float* __restrict__ C, int M, int N) {
  const int n0   = blockIdx.x * 64;
  const int m0   = blockIdx.y * 16;
  const int lane = threadIdx.x & 31;
  const int half = lane >> 4;   // 0: lanes 0-15, 1: lanes 16-31
  const int r    = lane & 15;

  const __bf16* arow  = A + (size_t)(m0 + r) * K;
  const __bf16* brow0 = B + (size_t)(n0 +  0 + r) * K;
  const __bf16* brow1 = B + (size_t)(n0 + 16 + r) * K;
  const __bf16* brow2 = B + (size_t)(n0 + 32 + r) * K;
  const __bf16* brow3 = B + (size_t)(n0 + 48 + r) * K;

  // A 16x32 bf16 lane layout: lanes 0-15 K={0..7,16..23}, lanes 16-31 K={8..15,24..31}
  auto loadA = [&](int k) {
    v16bf a;
    ((v8bf*)&a)[0] = *(const v8bf*)(arow + k + 8 * half);
    ((v8bf*)&a)[1] = *(const v8bf*)(arow + k + 16 + 8 * half);
    return a;
  };
  // B 32x16 bf16 lane layout: lanes 0-15 K={0..7,16..23}, lanes 16-31 K={8..15,24..31}
  auto loadB = [&](const __bf16* brow, int k) {
    v16bf b;
    ((v8bf*)&b)[0] = *(const v8bf*)(brow + k + 8 * half);
    ((v8bf*)&b)[1] = *(const v8bf*)(brow + k + 16 + 8 * half);
    return b;
  };

  v8f acc[4] = {};
  v16bf a_c  = loadA(0);
  v16bf b_c0 = loadB(brow0, 0), b_c1 = loadB(brow1, 0);
  v16bf b_c2 = loadB(brow2, 0), b_c3 = loadB(brow3, 0);

#pragma unroll
  for (int k = 0; k < K; k += 32) {
    v16bf a_n = a_c, b_n0 = b_c0, b_n1 = b_c1, b_n2 = b_c2, b_n3 = b_c3;
    if (k + 32 < K) { // prefetch next k-step's fragments before this step's WMMAs
      a_n  = loadA(k + 32);
      b_n0 = loadB(brow0, k + 32);
      b_n1 = loadB(brow1, k + 32);
      b_n2 = loadB(brow2, k + 32);
      b_n3 = loadB(brow3, k + 32);
    }
    acc[0] = __builtin_amdgcn_wmma_f32_16x16x32_bf16(false, a_c, false, b_c0, (short)0, acc[0], false, false);
    acc[1] = __builtin_amdgcn_wmma_f32_16x16x32_bf16(false, a_c, false, b_c1, (short)0, acc[1], false, false);
    acc[2] = __builtin_amdgcn_wmma_f32_16x16x32_bf16(false, a_c, false, b_c2, (short)0, acc[2], false, false);
    acc[3] = __builtin_amdgcn_wmma_f32_16x16x32_bf16(false, a_c, false, b_c3, (short)0, acc[3], false, false);
    a_c = a_n; b_c0 = b_n0; b_c1 = b_n1; b_c2 = b_n2; b_c3 = b_n3;
  }

  // C layout: VGPR j -> row M = m0 + j + 8*half, col N = n0 + nn*16 + (lane&15)
#pragma unroll
  for (int nn = 0; nn < 4; ++nn) {
    const int col  = n0 + nn * 16 + r;
    const float bv = (col < nsplit) ? bias0[col] : bias1[col - nsplit];
#pragma unroll
    for (int j = 0; j < 8; ++j)
      C[(size_t)(m0 + 8 * half + j) * N + col] = acc[nn][j] + bv;
  }
}

// -------------------- persistent recurrent GRU (2 independent directions) --------------------
// grid = 2 (blockIdx.x: 0 = forward cell, 1 = backward cell). 768 threads: one gate row each.
// Gi: [L, 1536] precomputed input gates (cols 0..767 = f cell, 768..1535 = b cell).
// Gi[t+1] is double-buffered into LDS with async-DMA during step t's dot product so the
// update phase (on the serial critical path) never touches global memory.
__global__ __launch_bounds__(768)
void k_gru_rec(const float* __restrict__ Gi,
               const __bf16* __restrict__ WhhF, const __bf16* __restrict__ WhhB,
               const float* __restrict__ bhhF, const float* __restrict__ bhhB,
               const float* __restrict__ h0F, const float* __restrict__ h0B,
               __bf16* __restrict__ Obf,  // [L,512] bf16 outputs (or null)
               float*  __restrict__ Of,   // [L,512] f32 outputs (or null)
               float* __restrict__ hFinF, float* __restrict__ hFinB, int L) {
  const int cell = blockIdx.x;
  const __bf16* Whh = cell ? WhhB : WhhF;
  const float*  bhh = cell ? bhhB : bhhF;
  const float*  h0  = cell ? h0B : h0F;
  float* hFin       = cell ? hFinB : hFinF;
  const int tid = threadIdx.x; // 0..767, one gate row

  __shared__ float hs[OUT_H];
  __shared__ float gh[3 * OUT_H];
  __shared__ float gi_buf[2][3 * OUT_H];
  if (tid < OUT_H) hs[tid] = h0[tid];
  const float bh = bhh[tid];
  const __bf16* wr = Whh + (size_t)tid * OUT_H;
  const float* gcol = Gi + cell * 768 + tid; // this thread's gate column
  // preload Gi[0] into buffer 0 (async; retired by the first in-loop wait)
  async_copy_b32(&gi_buf[0][tid], gcol);
  __syncthreads();

  for (int t = 0; t < L; ++t) {
    // kick off async prefetch of Gi[t+1] while we crunch the hidden matvec
    const bool pf = (t + 1 < L);
    if (pf) async_copy_b32(&gi_buf[(t + 1) & 1][tid], gcol + (size_t)(t + 1) * 1536);

    float s = bh;
#pragma unroll 8
    for (int k = 0; k < OUT_H; ++k) s += (float)wr[k] * hs[k];
    gh[tid] = s;

    // async loads complete in order: <=1 outstanding means Gi[t]'s copy has landed
    if (pf) asm volatile("s_wait_asynccnt 1" ::: "memory");
    else    asm volatile("s_wait_asynccnt 0" ::: "memory");
    __syncthreads();

    if (tid < OUT_H) {
      const float* gi = gi_buf[t & 1];
      float r = sigm(gi[tid] + gh[tid]);
      float z = sigm(gi[OUT_H + tid] + gh[OUT_H + tid]);
      float n = tanhf(gi[2 * OUT_H + tid] + r * gh[2 * OUT_H + tid]);
      float h = (1.0f - z) * n + z * hs[tid];
      hs[tid] = h;
      const size_t oi = (size_t)t * 512 + cell * OUT_H + tid;
      if (Obf) Obf[oi] = (__bf16)h;
      if (Of)  Of[oi]  = h;
    }
    __syncthreads();
  }
  if (tid < OUT_H && hFin) hFin[tid] = hs[tid];
}

// -------------------- row-wise log_softmax over 512 features --------------------
__global__ __launch_bounds__(256)
void k_logsoftmax_row(const float* __restrict__ H, float* __restrict__ out) {
  const int row = blockIdx.x, tid = threadIdx.x;
  const float* src = H + (size_t)row * DE_H;
  __shared__ float scr[256];
  float a = src[tid], b = src[tid + 256];
  scr[tid] = fmaxf(a, b);
  __syncthreads();
  for (int w = 128; w > 0; w >>= 1) { if (tid < w) scr[tid] = fmaxf(scr[tid], scr[tid + w]); __syncthreads(); }
  float mx = scr[0];
  __syncthreads();
  scr[tid] = expf(a - mx) + expf(b - mx);
  __syncthreads();
  for (int w = 128; w > 0; w >>= 1) { if (tid < w) scr[tid] += scr[tid + w]; __syncthreads(); }
  float lz = logf(scr[0]) + mx;
  out[(size_t)row * DE_H + tid]       = a - lz;
  out[(size_t)row * DE_H + tid + 256] = b - lz;
}

// -------------------- decoder precompute for the 6 possible input tokens --------------------
// dh is constant in the reference's greedy loop, so dec_step is a pure function of the token.
__global__ __launch_bounds__(1024)
void k_decoder(const float* __restrict__ dh, const float* __restrict__ eout,
               const float* __restrict__ de_emb,
               const float* __restrict__ attn_W, const float* __restrict__ attn_b,
               const float* __restrict__ comb_W, const float* __restrict__ comb_b,
               const float* __restrict__ de_Wih, const float* __restrict__ de_Whh,
               const float* __restrict__ de_bih, const float* __restrict__ de_bhh,
               const float* __restrict__ h2t_W,  const float* __restrict__ h2t_b,
               int* __restrict__ next_tok, float* __restrict__ logp0) {
  const int tid = threadIdx.x; // 0..1023
  __shared__ float dhs[DE_H];
  __shared__ float vbuf[2 * DE_H];
  __shared__ float aw[MAXLEN];
  __shared__ float scr[MAXLEN];
  __shared__ float obuf[DE_H];
  __shared__ float hnbuf[DE_H];
  __shared__ float gibuf[3 * DE_H];
  __shared__ float ghbuf[3 * DE_H];
  __shared__ float lg[TAGS];

  if (tid < DE_H) dhs[tid] = dh[tid];
  __syncthreads();

  for (int tok = 0; tok < TAGS; ++tok) {
    vbuf[tid] = (tid < DE_H) ? de_emb[tok * DE_H + tid] : dhs[tid - DE_H];
    __syncthreads();
    // attention scores + softmax over MAXLEN
    float s = attn_b[tid];
    const float* wr = attn_W + (size_t)tid * (2 * DE_H);
    for (int k = 0; k < 2 * DE_H; ++k) s += wr[k] * vbuf[k];
    scr[tid] = s;
    __syncthreads();
    for (int w = 512; w > 0; w >>= 1) { if (tid < w) scr[tid] = fmaxf(scr[tid], scr[tid + w]); __syncthreads(); }
    float mx = scr[0];
    __syncthreads();
    float ex = expf(s - mx);
    scr[tid] = ex;
    __syncthreads();
    for (int w = 512; w > 0; w >>= 1) { if (tid < w) scr[tid] += scr[tid + w]; __syncthreads(); }
    float inv = 1.0f / scr[0];
    __syncthreads();
    aw[tid] = ex * inv;
    __syncthreads();
    // ctx = aw @ eoutputs -> overwrite vbuf[DE_H..) (e half stays intact)
    if (tid < DE_H) {
      float c2 = 0.0f;
      for (int i = 0; i < MAXLEN; ++i) c2 += aw[i] * eout[(size_t)i * DE_H + tid];
      vbuf[DE_H + tid] = c2;
    }
    __syncthreads();
    // o = relu(comb_W @ cat(e, ctx) + comb_b)
    if (tid < DE_H) {
      float o = comb_b[tid];
      const float* cr = comb_W + (size_t)tid * (2 * DE_H);
      for (int k = 0; k < 2 * DE_H; ++k) o += cr[k] * vbuf[k];
      obuf[tid] = fmaxf(o, 0.0f);
    }
    __syncthreads();
    // GRU gate matvecs (1536 rows over 1024 threads)
    for (int row = tid; row < 3 * DE_H; row += 1024) {
      float gi = de_bih[row], ghv = de_bhh[row];
      const float* wi = de_Wih + (size_t)row * DE_H;
      const float* wh = de_Whh + (size_t)row * DE_H;
      for (int k = 0; k < DE_H; ++k) { gi += wi[k] * obuf[k]; ghv += wh[k] * dhs[k]; }
      gibuf[row] = gi; ghbuf[row] = ghv;
    }
    __syncthreads();
    if (tid < DE_H) {
      float r = sigm(gibuf[tid] + ghbuf[tid]);
      float z = sigm(gibuf[DE_H + tid] + ghbuf[DE_H + tid]);
      float n = tanhf(gibuf[2 * DE_H + tid] + r * ghbuf[2 * DE_H + tid]);
      hnbuf[tid] = (1.0f - z) * n + z * dhs[tid];
    }
    __syncthreads();
    if (tid < TAGS) {
      float s2 = h2t_b[tid];
      const float* hr = h2t_W + (size_t)tid * DE_H;
      for (int k = 0; k < DE_H; ++k) s2 += hr[k] * hnbuf[k];
      lg[tid] = s2;
    }
    __syncthreads();
    if (tid == 0) {
      float mx2 = lg[0]; int am = 0;
      for (int k = 1; k < TAGS; ++k) if (lg[k] > mx2) { mx2 = lg[k]; am = k; }
      float se = 0.0f;
      for (int k = 0; k < TAGS; ++k) se += expf(lg[k] - mx2);
      logp0[tok] = lg[0] - mx2 - logf(se);
      next_tok[tok] = am;
    }
    __syncthreads();
  }
}

// -------------------- 6-state Markov chain + h2t row sums --------------------
__global__ void k_markov(const int* __restrict__ nt, const float* __restrict__ lp0,
                         const float* __restrict__ h2t_W,
                         float* __restrict__ c, float* __restrict__ rs, int L) {
  const int tid = threadIdx.x;
  if (tid == 0) {
    int cur = START_TAG;
    for (int t = 0; t < L; ++t) { c[t] = lp0[cur]; cur = nt[cur]; }
  }
  if (tid < TAGS) {
    float s = 0.0f;
    for (int j = 0; j < DE_H; ++j) s += h2t_W[tid * DE_H + j];
    rs[tid] = s;
  }
}

// -------------------- rank-1 output --------------------
__global__ void k_final(const float* __restrict__ c, const float* __restrict__ rs,
                        const float* __restrict__ h2t_b, float* __restrict__ out, int n) {
  int i = blockIdx.x * blockDim.x + threadIdx.x;
  if (i < n) {
    int t = i / TAGS, k = i % TAGS;
    out[i] = c[t] * rs[k] + h2t_b[k];
  }
}

// ==================== host launcher ====================
extern "C" void kernel_launch(void* const* d_in, const int* in_sizes, int n_in,
                              void* d_out, int out_size, void* d_ws, size_t ws_size,
                              hipStream_t stream) {
  const int*   x    = (const int*)  d_in[0];
  const float* emb  = (const float*)d_in[1];
  const float* W0f  = (const float*)d_in[2];
  const float* U0f  = (const float*)d_in[3];
  const float* bi0f = (const float*)d_in[4];
  const float* bh0f = (const float*)d_in[5];
  const float* W0b  = (const float*)d_in[6];
  const float* U0b  = (const float*)d_in[7];
  const float* bi0b = (const float*)d_in[8];
  const float* bh0b = (const float*)d_in[9];
  const float* W1f  = (const float*)d_in[10];
  const float* U1f  = (const float*)d_in[11];
  const float* bi1f = (const float*)d_in[12];
  const float* bh1f = (const float*)d_in[13];
  const float* W1b  = (const float*)d_in[14];
  const float* U1b  = (const float*)d_in[15];
  const float* bi1b = (const float*)d_in[16];
  const float* bh1b = (const float*)d_in[17];
  const float* eh0  = (const float*)d_in[18];
  const float* de_emb = (const float*)d_in[19];
  const float* attn_W = (const float*)d_in[20];
  const float* attn_b = (const float*)d_in[21];
  const float* comb_W = (const float*)d_in[22];
  const float* comb_b = (const float*)d_in[23];
  const float* de_Wih = (const float*)d_in[24];
  const float* de_Whh = (const float*)d_in[25];
  const float* de_bih = (const float*)d_in[26];
  const float* de_bhh = (const float*)d_in[27];
  const float* h2t_W  = (const float*)d_in[28];
  const float* h2t_b  = (const float*)d_in[29];

  const int L = in_sizes[0]; // 1024 (== MAXLEN)

  // workspace carve-out (256B aligned)
  char* p = (char*)d_ws;
  auto carve = [&](size_t bytes) { char* r = p; p += (bytes + 255) & ~(size_t)255; return r; };
  __bf16* Xbf   = (__bf16*)carve((size_t)L * IN_H * 2);
  __bf16* W0bf  = (__bf16*)carve((size_t)1536 * 256 * 2); // [W0f;W0b]
  __bf16* U0bf  = (__bf16*)carve((size_t)1536 * 256 * 2); // [U0f;U0b]
  __bf16* W1bf  = (__bf16*)carve((size_t)1536 * 512 * 2); // [W1f;W1b]
  __bf16* U1bf  = (__bf16*)carve((size_t)1536 * 256 * 2); // [U1f;U1b]
  float*  Gi0   = (float*) carve((size_t)L * 1536 * 4);
  float*  Gi1   = (float*) carve((size_t)L * 1536 * 4);
  __bf16* O0bf  = (__bf16*)carve((size_t)L * 512 * 2);
  float*  H1    = (float*) carve((size_t)L * 512 * 4);
  float*  eout  = (float*) carve((size_t)MAXLEN * 512 * 4);
  float*  dh    = (float*) carve(512 * 4);
  float*  lp0   = (float*) carve(64);
  int*    ntok  = (int*)   carve(64);
  float*  cbuf  = (float*) carve((size_t)L * 4);
  float*  rs    = (float*) carve(64);

  const int T = 256;
  auto nb = [](int n, int t) { return (n + t - 1) / t; };

  // stage: embeddings + weight conversion to bf16 (stacked f/b)
  k_embed<<<nb(L * IN_H, T), T, 0, stream>>>(x, emb, Xbf, L);
  k_f32_to_bf16<<<nb(768 * 256, T), T, 0, stream>>>(W0f, W0bf,             768 * 256);
  k_f32_to_bf16<<<nb(768 * 256, T), T, 0, stream>>>(W0b, W0bf + 768 * 256, 768 * 256);
  k_f32_to_bf16<<<nb(768 * 256, T), T, 0, stream>>>(U0f, U0bf,             768 * 256);
  k_f32_to_bf16<<<nb(768 * 256, T), T, 0, stream>>>(U0b, U0bf + 768 * 256, 768 * 256);
  k_f32_to_bf16<<<nb(768 * 512, T), T, 0, stream>>>(W1f, W1bf,             768 * 512);
  k_f32_to_bf16<<<nb(768 * 512, T), T, 0, stream>>>(W1b, W1bf + 768 * 512, 768 * 512);
  k_f32_to_bf16<<<nb(768 * 256, T), T, 0, stream>>>(U1f, U1bf,             768 * 256);
  k_f32_to_bf16<<<nb(768 * 256, T), T, 0, stream>>>(U1b, U1bf + 768 * 256, 768 * 256);
  k_fill0<<<nb(MAXLEN * 512, T), T, 0, stream>>>(eout, MAXLEN * 512);

  // encoder layer 0: batched input gates (WMMA, 16x64 tiles) + sequential hidden recurrence
  dim3 g0(1536 / 64, L / 16);
  k_wmma_gemm<256><<<g0, 32, 0, stream>>>(Xbf, W0bf, bi0f, bi0b, 768, Gi0, L, 1536);
  k_gru_rec<<<2, 768, 0, stream>>>(Gi0, U0bf, U0bf + 768 * 256, bh0f, bh0b,
                                   eh0 + 0, eh0 + 256, O0bf, nullptr, dh, nullptr, L);

  // encoder layer 1: batched input gates over layer-0 outputs (WMMA) + recurrence
  k_wmma_gemm<512><<<g0, 32, 0, stream>>>(O0bf, W1bf, bi1f, bi1b, 768, Gi1, L, 1536);
  k_gru_rec<<<2, 768, 0, stream>>>(Gi1, U1bf, U1bf + 768 * 256, bh1f, bh1b,
                                   eh0 + 512, eh0 + 768, nullptr, H1, nullptr, dh + 256, L);

  // eoutputs = row-wise log_softmax of concat(h1f,h1b)
  k_logsoftmax_row<<<L, 256, 0, stream>>>(H1, eout);

  // decoder: evaluate dec_step for all 6 possible tokens, then iterate the chain
  k_decoder<<<1, 1024, 0, stream>>>(dh, eout, de_emb, attn_W, attn_b, comb_W, comb_b,
                                    de_Wih, de_Whh, de_bih, de_bhh, h2t_W, h2t_b,
                                    ntok, lp0);
  k_markov<<<1, 32, 0, stream>>>(ntok, lp0, h2t_W, cbuf, rs, L);
  k_final<<<nb(L * TAGS, T), T, 0, stream>>>(cbuf, rs, h2t_b, (float*)d_out, L * TAGS);
}
